// CitationGAT_67104569033191
// MI455X (gfx1250) — matrix operand order repaired and checked by reading, not compiled, not run on hardware
//
#include <hip/hip_runtime.h>

typedef __bf16 bf16;
typedef __bf16 bf16x16 __attribute__((ext_vector_type(16)));
typedef __bf16 bf16x2  __attribute__((ext_vector_type(2)));
typedef float  f32x8   __attribute__((ext_vector_type(8)));

#define LRELU_ALPHA 0.2f
#define NEGINF -9.0e15f

constexpr int Bsz  = 4;
constexpr int Nn   = 2048;
constexpr int FIN  = 512;
constexpr int Hh   = 128;
constexpr int NH   = 4;
constexpr int FOUT = 256;

// ---- CDNA5 async global->LDS copy (guarded; falls back to sync copy) -------
#ifndef __has_builtin
#define __has_builtin(x) 0
#endif
#if __has_builtin(__builtin_amdgcn_global_load_async_to_lds_b128)
#define HAVE_ASYNC_LDS 1
#else
#define HAVE_ASYNC_LDS 0
#endif

typedef int v4i __attribute__((vector_size(4 * sizeof(int))));
typedef __attribute__((address_space(1))) v4i* gv4i_p;   // global int4*
typedef __attribute__((address_space(3))) v4i* lv4i_p;   // LDS int4*

__device__ __forceinline__ void async_copy16(void* l, const void* g) {
#if HAVE_ASYNC_LDS
    __builtin_amdgcn_global_load_async_to_lds_b128((gv4i_p)g, (lv4i_p)l, 0, 0);
#else
    *(uint4*)l = *(const uint4*)g;
#endif
}

#if HAVE_ASYNC_LDS
#if __has_builtin(__builtin_amdgcn_s_wait_asynccnt)
#define ASYNC_WAIT(n) __builtin_amdgcn_s_wait_asynccnt(n)
#else
#define ASYNC_WAIT(n) asm volatile("s_wait_asynccnt %0" ::"i"(n) : "memory")
#endif
#else
#define ASYNC_WAIT(n) ((void)0)
#endif

__device__ __forceinline__ unsigned packbf(float a, float b) {
    bf16x2 v; v[0] = (bf16)a; v[1] = (bf16)b;
    return __builtin_bit_cast(unsigned, v);
}

// ---------------------------------------------------------------------------
__global__ void f2bf_kernel(const float* __restrict__ s, bf16* __restrict__ d, int n) {
    int i = blockIdx.x * 256 + threadIdx.x;
    if (i < n) d[i] = (bf16)s[i];
}

// pack f32 weights [nmat][K][ncols] -> WMMA-B-ready uint32 layout:
// dst[mat][(jblk*ncols + col)*16 + kk] = pair(W[k=jblk*32+2kk][col], W[k+1][col])
__global__ void packw_kernel(const float* __restrict__ src, unsigned* __restrict__ dst,
                             int K, int ncols, int nmat) {
    int o = blockIdx.x * 256 + threadIdx.x;
    int per = (K / 2) * ncols;
    if (o >= nmat * per) return;
    int mat = o / per, o2 = o % per;
    int kk = o2 & 15;
    int colj = o2 >> 4;
    int col = colj % ncols;
    int jblk = colj / ncols;
    int k = jblk * 32 + 2 * kk;
    const float* s = src + (size_t)mat * K * ncols + (size_t)k * ncols + col;
    dst[o] = packbf(s[0], s[ncols]);
}

__global__ void packadj_kernel(const int* __restrict__ adj, unsigned* __restrict__ adjp, int nwords) {
    int i = blockIdx.x * 256 + threadIdx.x;
    if (i >= nwords) return;
    const int* p = adj + (size_t)i * 32;
    unsigned w = 0;
#pragma unroll
    for (int t = 0; t < 32; ++t) w |= (p[t] > 0 ? 1u : 0u) << t;
    adjp[i] = w;
}

// ---------------------------------------------------------------------------
// Tiled bf16 WMMA GEMM, double-buffered async LDS staging.
// Cf[M,ldc] = A[M,K] * W ; CbT = bf16 result in K-pair-packed B-operand layout.
template <bool HEADW>
__global__ __launch_bounds__(256) void gemm_kernel(
    const bf16* __restrict__ A, const unsigned* __restrict__ WT,
    float* __restrict__ Cf, unsigned* __restrict__ CbT,
    int K, int ldc, int ldwN) {
    __shared__ bf16     As[2][128][32];   // fragment-ordered chunks
    __shared__ unsigned Bt[2][64][16];    // [col][kk] pair-packed
    const int tid = threadIdx.x, lane = tid & 31, wv = tid >> 5;
    const int colbase = blockIdx.x * 64;
    const size_t rowbase = (size_t)blockIdx.y * 128;

    f32x8 acc[4];
#pragma unroll
    for (int c = 0; c < 4; ++c) acc[c] = (f32x8)0.f;

    const int r0 = tid >> 2, c80 = tid & 3;
    const int dc0 = c80 * 8;     // contiguous chunks 0,8,16,24
    const int r1 = (tid + 256) >> 2;

    auto stage = [&](int kb, int buf) {
        int k0 = kb << 5;
        async_copy16(&As[buf][r0][dc0], &A[(rowbase + r0) * K + k0 + c80 * 8]);
        async_copy16(&As[buf][r1][dc0], &A[(rowbase + r1) * K + k0 + c80 * 8]);
        size_t base;
        if (HEADW) {
            int head = colbase >> 7, colw = colbase & (Hh - 1);
            base = ((size_t)(head * 16 + kb) * Hh + colw) * 16;
        } else {
            base = ((size_t)kb * ldwN + colbase) * 16;
        }
        async_copy16(((uint4*)&Bt[buf][0][0]) + tid, ((const uint4*)(WT + base)) + tid);
    };

    const int nk = K >> 5;
    stage(0, 0);
    const int am = lane & 15, bn = lane & 15, kkb = (lane < 16) ? 0 : 8;
    for (int kb = 0; kb < nk; ++kb) {
        if (kb + 1 < nk) { stage(kb + 1, (kb + 1) & 1); ASYNC_WAIT(3); }
        else             { ASYNC_WAIT(0); }
        __syncthreads();
        const int bsel = kb & 1;
        bf16x16 af = *(const bf16x16*)&As[bsel][wv * 16 + am][(lane >> 4) << 4];
        bf16x16 bfv[4];
#pragma unroll
        for (int c = 0; c < 4; ++c) bfv[c] = *(const bf16x16*)&Bt[bsel][c * 16 + bn][kkb];
#pragma unroll
        for (int c = 0; c < 4; ++c)
            acc[c] = __builtin_amdgcn_wmma_f32_16x16x32_bf16(false, af, false, bfv[c],
                                                             (short)0, acc[c], false, false);
        __syncthreads();
    }

    const int m0 = (lane < 16) ? 0 : 8, coll = lane & 15;
#pragma unroll
    for (int c = 0; c < 4; ++c) {
        int col = colbase + c * 16 + coll;
#pragma unroll
        for (int t = 0; t < 4; ++t) {
            size_t grow = rowbase + wv * 16 + m0 + 2 * t;
            float v0 = acc[c][2 * t], v1 = acc[c][2 * t + 1];
            Cf[grow * ldc + col] = v0;
            Cf[(grow + 1) * ldc + col] = v1;
            CbT[((grow >> 5) * ldc + col) * 16 + ((grow >> 1) & 15)] = packbf(v0, v1);
        }
    }
}

// ---------------------------------------------------------------------------
__global__ __launch_bounds__(256) void fdot_kernel(
    const float* __restrict__ src, int ld, int ncols, int nHeads,
    const float* __restrict__ a, float* __restrict__ f1, float* __restrict__ f2) {
    int gw = (blockIdx.x * 256 + threadIdx.x) >> 5;
    int lane = threadIdx.x & 31;
    if (gw >= nHeads * Bsz * Nn) return;
    int h = gw / (Bsz * Nn);
    int bn = gw % (Bsz * Nn);
    const float* row = src + (size_t)bn * ld + h * ncols;
    const float* a1 = a + h * 2 * ncols;
    const float* a2 = a1 + ncols;
    float s1 = 0.f, s2 = 0.f;
    for (int c = lane; c < ncols; c += 32) { float v = row[c]; s1 += v * a1[c]; s2 += v * a2[c]; }
    for (int off = 16; off; off >>= 1) { s1 += __shfl_xor(s1, off, 32); s2 += __shfl_xor(s2, off, 32); }
    if (lane == 0) { f1[gw] = s1; f2[gw] = s2; }
}

// ---------------------------------------------------------------------------
// Fused GAT attention with double-buffered async Wh tiles.
// Softmax shift m_i = lrelu(f1_i + max_j f2_j) (upper bound; shift-invariant).
template <bool F32OUT>
__global__ __launch_bounds__(256) void attn_kernel(
    const unsigned* __restrict__ WhT, int ldw,
    const float* __restrict__ f1, const float* __restrict__ f2, int fYstride,
    const unsigned* __restrict__ adjp, const float* __restrict__ mask,
    float* __restrict__ outF, bf16* __restrict__ outB, int ldo) {
    __shared__ float    f2s[Nn];
    __shared__ float    f1s[128];
    __shared__ float    red[256];
    __shared__ unsigned Bt[2][128][16];   // [col][kk] pair-packed Wh j-tile
    __shared__ bf16     ps[8][16][32];    // fragment-ordered p rows

    const int b = blockIdx.z, y = blockIdx.y;
    const int tid = threadIdx.x, lane = tid & 31;
    const int wv = __builtin_amdgcn_readfirstlane(tid >> 5);   // scalar wave id
    const int row0 = blockIdx.x * 128;
    const float* f1p = f1 + (size_t)y * fYstride + (size_t)b * Nn;
    const float* f2p = f2 + (size_t)y * fYstride + (size_t)b * Nn;

    // async stage f2 (2048 f32) and f1 (128 f32)
    async_copy16(((uint4*)f2s) + tid, ((const uint4*)f2p) + tid);
    async_copy16(((uint4*)f2s) + tid + 256, ((const uint4*)f2p) + tid + 256);
    if (tid < 32) async_copy16(((uint4*)f1s) + tid, ((const uint4*)(f1p + row0)) + tid);
    ASYNC_WAIT(0);
    __syncthreads();

    // block max of f2
    float mv = -3.0e38f;
    for (int i = tid; i < Nn; i += 256) mv = fmaxf(mv, f2s[i]);
    red[tid] = mv;
    __syncthreads();
    if (tid < 128) red[tid] = fmaxf(red[tid], red[tid + 128]);
    __syncthreads();
    if (tid < 32) {
        float m = fmaxf(red[tid], fmaxf(red[tid + 32], fmaxf(red[tid + 64], red[tid + 96])));
        for (int off = 16; off; off >>= 1) m = fmaxf(m, __shfl_xor(m, off, 32));
        red[tid] = m;
    }
    __syncthreads();
    const float maxf2 = red[0];

    float mrow[16];
#pragma unroll
    for (int r = 0; r < 16; ++r) {
        float s = f1s[wv * 16 + r] + maxf2;
        mrow[r] = s > 0.f ? s : LRELU_ALPHA * s;
    }

    f32x8 acc[8];
#pragma unroll
    for (int c = 0; c < 8; ++c) acc[c] = (f32x8)0.f;
    float lsum[16];
#pragma unroll
    for (int r = 0; r < 16; ++r) lsum[r] = 0.f;

    const int pidx = lane;
    const unsigned* WhBase = WhT + (size_t)b * (Nn / 32) * ldw * 16 + (size_t)y * 128 * 16;
    const size_t adjbase = ((size_t)b * Nn + row0 + wv * 16) * 64;  // scalar (wv uniform)

    {   // prologue: stage tile 0
        const uint4* wsrc = (const uint4*)WhBase;
        async_copy16(((uint4*)&Bt[0][0][0]) + tid, wsrc + tid);
        async_copy16(((uint4*)&Bt[0][0][0]) + tid + 256, wsrc + tid + 256);
    }

    const int bn = lane & 15, kkb = (lane < 16) ? 0 : 8;
    for (int js = 0; js < 64; ++js) {
        if (js + 1 < 64) {  // stream next tile while computing this one
            const uint4* wsrc = (const uint4*)(WhBase + (size_t)(js + 1) * ldw * 16);
            uint4* dst = (uint4*)&Bt[(js + 1) & 1][0][0];
            async_copy16(dst + tid, wsrc + tid);
            async_copy16(dst + tid + 256, wsrc + tid + 256);
            ASYNC_WAIT(2);
        } else {
            ASYNC_WAIT(0);
        }
        __syncthreads();

        const float f2v = f2s[js * 32 + lane];
#pragma unroll
        for (int r = 0; r < 16; ++r) {
            float s = f1s[wv * 16 + r] + f2v;
            s = s > 0.f ? s : LRELU_ALPHA * s;
            s = ((adjp[adjbase + (size_t)r * 64 + js] >> lane) & 1u) ? s : NEGINF;
            float p = __expf(s - mrow[r]);
            lsum[r] += p;
            ps[wv][r][pidx] = (bf16)p;
        }
        bf16x16 af = *(const bf16x16*)&ps[wv][lane & 15][(lane >> 4) << 4];
        const int bsel = js & 1;
        bf16x16 bfv[4];
#pragma unroll
        for (int c = 0; c < 4; ++c) bfv[c] = *(const bf16x16*)&Bt[bsel][c * 16 + bn][kkb];
#pragma unroll
        for (int c = 0; c < 4; ++c)
            acc[c] = __builtin_amdgcn_wmma_f32_16x16x32_bf16(false, af, false, bfv[c],
                                                             (short)0, acc[c], false, false);
#pragma unroll
        for (int c = 0; c < 4; ++c) bfv[c] = *(const bf16x16*)&Bt[bsel][(c + 4) * 16 + bn][kkb];
#pragma unroll
        for (int c = 0; c < 4; ++c)
            acc[c + 4] = __builtin_amdgcn_wmma_f32_16x16x32_bf16(false, af, false, bfv[c],
                                                                 (short)0, acc[c + 4], false, false);
        __syncthreads();
    }

#pragma unroll
    for (int r = 0; r < 16; ++r)
        for (int off = 16; off; off >>= 1) lsum[r] += __shfl_xor(lsum[r], off, 32);

    const int m0 = (lane < 16) ? 0 : 8, coll = lane & 15;
#pragma unroll
    for (int c = 0; c < 8; ++c)
#pragma unroll
        for (int i = 0; i < 8; ++i) {
            float li = (lane < 16) ? lsum[i] : lsum[8 + i];
            float v = acc[c][i] / li;
            size_t gr = (size_t)b * Nn + row0 + wv * 16 + m0 + i;
            int col = y * 128 + c * 16 + coll;
            if constexpr (F32OUT) {
                v *= mask[gr];
                outF[gr * ldo + col] = v;
            } else {
                outB[gr * ldo + col] = (bf16)(v > 0.f ? v : __expf(v) - 1.f);
            }
        }
}

// ---------------------------------------------------------------------------
__global__ __launch_bounds__(256) void pool_kernel(
    const float* __restrict__ xout, const float* __restrict__ mask, float* __restrict__ pooled) {
    int b = blockIdx.x, tid = threadIdx.x;
    __shared__ float msh[256];
    float ms = 0.f;
    for (int n = tid; n < Nn; n += 256) ms += mask[(size_t)b * Nn + n];
    msh[tid] = ms; __syncthreads();
    for (int s = 128; s; s >>= 1) { if (tid < s) msh[tid] += msh[tid + s]; __syncthreads(); }
    float denom = msh[0] + 1e-10f;
    float acc = 0.f;
    for (int n = 0; n < Nn; ++n) acc += xout[((size_t)b * Nn + n) * FOUT + tid];
    pooled[b * FOUT + tid] = acc / denom;
}

__global__ __launch_bounds__(256) void head_kernel(
    const float* __restrict__ pooled, const float* __restrict__ Wp,
    const float* __restrict__ bp, float* __restrict__ out) {
    int i = blockIdx.x * 256 + threadIdx.x;   // 1024
    int b = i >> 8, fo = i & 255;
    float acc = bp[fo];
    for (int f = 0; f < 256; ++f) acc += pooled[b * 256 + f] * Wp[f * 256 + fo];
    out[i] = fmaxf(acc, 0.f);
}

// ---------------------------------------------------------------------------
extern "C" void kernel_launch(void* const* d_in, const int* in_sizes, int n_in,
                              void* d_out, int out_size, void* d_ws, size_t ws_size,
                              hipStream_t stream) {
    (void)in_sizes; (void)n_in; (void)out_size; (void)ws_size;
    const float* x       = (const float*)d_in[0];
    const int*   adj     = (const int*)d_in[1];
    const float* mask    = (const float*)d_in[2];
    const float* W_heads = (const float*)d_in[3];
    const float* a_heads = (const float*)d_in[4];
    const float* W_out   = (const float*)d_in[5];
    const float* a_out   = (const float*)d_in[6];
    const float* W_pool  = (const float*)d_in[7];
    const float* b_pool  = (const float*)d_in[8];

    char* w = (char*)d_ws;
    size_t off = 0;
    auto alloc = [&](size_t bytes) { void* p = w + off; off += (bytes + 255) & ~(size_t)255; return p; };
    bf16*     x_bf   = (bf16*)alloc((size_t)Bsz * Nn * FIN * 2);
    unsigned* WhbT   = (unsigned*)alloc((size_t)NH * FIN * Hh * 2);
    unsigned* WobT   = (unsigned*)alloc((size_t)FIN * FOUT * 2);
    unsigned* adjp   = (unsigned*)alloc((size_t)Bsz * Nn * 64 * 4);
    float*    xW     = (float*)alloc((size_t)Bsz * Nn * FIN * 4);
    unsigned* xWT    = (unsigned*)alloc((size_t)Bsz * Nn * FIN * 2);
    float*    f1h    = (float*)alloc((size_t)NH * Bsz * Nn * 4);
    float*    f2h    = (float*)alloc((size_t)NH * Bsz * Nn * 4);
    bf16*     xcat   = (bf16*)alloc((size_t)Bsz * Nn * FIN * 2);
    float*    WhO    = (float*)alloc((size_t)Bsz * Nn * FOUT * 4);
    unsigned* WhOT   = (unsigned*)alloc((size_t)Bsz * Nn * FOUT * 2);
    float*    f1o    = (float*)alloc((size_t)Bsz * Nn * 4);
    float*    f2o    = (float*)alloc((size_t)Bsz * Nn * 4);
    float*    pooled = (float*)alloc((size_t)Bsz * FOUT * 4);

    float* outEmb  = (float*)d_out;
    float* outXout = outEmb + Bsz * FOUT;    // x_out at d_out + 1024

    int nx = Bsz * Nn * FIN;
    f2bf_kernel<<<(nx + 255) / 256, 256, 0, stream>>>(x, x_bf, nx);
    int nwhw = NH * (FIN / 2) * Hh;
    packw_kernel<<<(nwhw + 255) / 256, 256, 0, stream>>>(W_heads, WhbT, FIN, Hh, NH);
    int nwow = (FIN / 2) * FOUT;
    packw_kernel<<<(nwow + 255) / 256, 256, 0, stream>>>(W_out, WobT, FIN, FOUT, 1);
    int nwords = Bsz * Nn * 64;
    packadj_kernel<<<(nwords + 255) / 256, 256, 0, stream>>>(adj, adjp, nwords);

    gemm_kernel<true><<<dim3(FIN / 64, (Bsz * Nn) / 128), 256, 0, stream>>>(
        x_bf, WhbT, xW, xWT, FIN, FIN, 0);

    fdot_kernel<<<(NH * Bsz * Nn) / 8, 256, 0, stream>>>(xW, FIN, Hh, NH, a_heads, f1h, f2h);

    attn_kernel<false><<<dim3(Nn / 128, NH, Bsz), 256, 0, stream>>>(
        xWT, FIN, f1h, f2h, Bsz * Nn, adjp, nullptr, nullptr, xcat, FIN);

    gemm_kernel<false><<<dim3(FOUT / 64, (Bsz * Nn) / 128), 256, 0, stream>>>(
        xcat, WobT, WhO, WhOT, FIN, FOUT, FOUT);

    fdot_kernel<<<(Bsz * Nn) / 8, 256, 0, stream>>>(WhO, FOUT, FOUT, 1, a_out, f1o, f2o);

    attn_kernel<true><<<dim3(Nn / 128, FOUT / 128, Bsz), 256, 0, stream>>>(
        WhOT, FOUT, f1o, f2o, 0, adjp, mask, outXout, nullptr, FOUT);

    pool_kernel<<<Bsz, 256, 0, stream>>>(outXout, mask, pooled);
    head_kernel<<<(Bsz * FOUT) / 256, 256, 0, stream>>>(pooled, W_pool, b_pool, outEmb);
}
